// BiLSTM_CRF_40046275068213
// MI455X (gfx1250) — matrix operand order repaired and checked by reading, not compiled, not run on hardware
//
#include <hip/hip_runtime.h>
#include <hip/hip_bf16.h>
#include <math.h>

typedef __attribute__((ext_vector_type(16))) __bf16 v16bf;
typedef __attribute__((ext_vector_type(8)))  float  v8f;
typedef __attribute__((ext_vector_type(2)))  float  v2f;
typedef __attribute__((ext_vector_type(8)))  unsigned int v8u;
typedef unsigned short u16;
typedef unsigned int   u32;

// ---------- helpers ----------
__device__ __forceinline__ u32 bf16bits(float x){
  u32 u = __builtin_bit_cast(u32, x);
  return (u + 0x7FFFu + ((u >> 16) & 1u)) >> 16;   // RNE f32->bf16
}
__device__ __forceinline__ u16 f2bf(float x){ return (u16)bf16bits(x); }
__device__ __forceinline__ u32 pack2bf(float a, float b){ return bf16bits(a) | (bf16bits(b) << 16); }
__device__ __forceinline__ v8f v8f_bcast(float x){
  v8f r;
#pragma unroll
  for(int i=0;i<8;++i) r[i]=x;
  return r;
}
__device__ __forceinline__ float sigf(float x){ return 1.0f/(1.0f+__expf(-x)); }

// A-fragment (16x32 bf16): lane<16 holds K=[kb*32+0..7, +16..23], lane>=16 K=[+8..15, +24..31], row M=lane%16
__device__ __forceinline__ v16bf load_afrag(const u16* base, int ld, int kb, int lane){
  const u16* p = base + (size_t)(lane & 15) * ld + kb*32 + ((lane & 16) ? 8 : 0);
  uint4 x = *(const uint4*)p;
  uint4 y = *(const uint4*)(p + 16);
  v8u u = { x.x, x.y, x.z, x.w, y.x, y.y, y.z, y.w };
  return __builtin_bit_cast(v16bf, u);
}
// B-fragment (32x16 bf16) from W[n][k] row-major: lane N=lane%16, lane<16 K=[kb*32+0..7, +16..23], lane>=16 K=[+8..15, +24..31]
__device__ __forceinline__ v16bf load_bfrag(const u16* base, int ld, int kb, int lane){
  const u16* p = base + (size_t)(lane & 15) * ld + kb*32 + ((lane & 16) ? 8 : 0);
  uint4 x = *(const uint4*)p;
  uint4 y = *(const uint4*)(p + 16);
  v8u u = { x.x, x.y, x.z, x.w, y.x, y.y, y.z, y.w };
  return __builtin_bit_cast(v16bf, u);
}
__device__ __forceinline__ v8f wmma_bf(v16bf a, v16bf b, v8f c){
  return __builtin_amdgcn_wmma_f32_16x16x32_bf16(false, a, false, b, (short)0, c, false, false);
}
__device__ __forceinline__ v8f wmma_f32k4(v2f a, v2f b, v8f c){
  return __builtin_amdgcn_wmma_f32_16x16x4_f32(false, a, false, b, (short)0, c, false, false);
}

// ---------- weight conversion ----------
__global__ void cvt_bf16_kernel(const float* __restrict__ s, u16* __restrict__ d, int n){
  int i = blockIdx.x*blockDim.x + threadIdx.x;
  if(i < n) d[i] = f2bf(s[i]);
}
__global__ void pad_tagw_kernel(const float* __restrict__ w, u16* __restrict__ d){
  int i = blockIdx.x*blockDim.x + threadIdx.x;
  if(i < 32*704){
    int r = i/704, c = i%704;
    d[i] = (r < 27) ? f2bf(w[r*704 + c]) : (u16)0;
  }
}

// ---------- feature projection: fv = features^T @ feat_W^T + feat_b  (K=4 exactly -> f32 WMMA) ----------
__global__ __launch_bounds__(256,1) void feat_kernel(
    const float* __restrict__ features, const float* __restrict__ fW,
    const float* __restrict__ fb, float* __restrict__ fv)
{
  int tid = threadIdx.x, wave = tid>>5, lane = tid&31;
  int mt = blockIdx.x*8 + wave;                 // 1024 M-tiles
  int coll = lane & 15;
  int koff = (lane & 16) ? 2 : 0;
  int roff = (lane & 16) ? 8 : 0;
  int m = mt*16 + coll, b = m>>7, s = m&127;
  v2f a;
  a[0] = features[(size_t)(b*4 + koff  )*128 + s];
  a[1] = features[(size_t)(b*4 + koff+1)*128 + s];
#pragma unroll
  for(int nt=0; nt<8; ++nt){
    int n = nt*16 + coll;
    v2f bb;
    bb[0] = fW[n*4 + koff];
    bb[1] = fW[n*4 + koff + 1];
    v8f acc = v8f_bcast(fb[n]);
    acc = wmma_f32k4(a, bb, acc);
#pragma unroll
    for(int e=0;e<8;++e)
      fv[(size_t)(mt*16 + e + roff)*128 + n] = acc[e];
  }
}

// ---------- char BiLSTM + output projection -> chvec[16384][64] ----------
__global__ __launch_bounds__(128,1) void char_lstm_kernel(
    const int*  __restrict__ chars2ix, const float* __restrict__ char_emb,
    const u16* __restrict__ Wih_f, const u16* __restrict__ Whh_f,
    const float* __restrict__ bih_f, const float* __restrict__ bhh_f,
    const u16* __restrict__ Wih_b, const u16* __restrict__ Whh_b,
    const float* __restrict__ bih_b, const float* __restrict__ bhh_b,
    const u16* __restrict__ coutW, const float* __restrict__ cout_b,
    float* __restrict__ chvec)
{
  __shared__ u16 lds[4*4096];            // per wave: xt[16*64], h[16*64], hcat[16*128]
  int tid = threadIdx.x, wave = tid>>5, lane = tid&31;
  u16* xt  = lds + wave*4096;
  u16* hsh = xt + 1024;
  u16* hc  = xt + 2048;
  int word0 = blockIdx.x*64 + wave*16;
  int row = lane & 15, half = lane >> 4;

  for(int dir=0; dir<2; ++dir){
    const u16* Wih = dir ? Wih_b : Wih_f;
    const u16* Whh = dir ? Whh_b : Whh_f;
    const float* bih = dir ? bih_b : bih_f;
    const float* bhh = dir ? bhh_b : bhh_f;
    { // zero h state (bf16 0 == 0 bits)
      u32* hz = (u32*)hsh;
#pragma unroll
      for(int i=0;i<16;++i) hz[lane*16 + i] = 0u;
    }
    v8f cst[4];
#pragma unroll
    for(int p=0;p<4;++p) cst[p] = v8f_bcast(0.f);

    for(int step=0; step<16; ++step){
      int t = dir ? (15 - step) : step;
      // gather char embedding row -> bf16 tile in LDS
      int widx = word0 + row;
      int ci = chars2ix[widx*16 + t];
      const float2* er = (const float2*)(char_emb + (size_t)ci*64 + half*32);
      u32* xr = (u32*)(xt + row*64 + half*32);
#pragma unroll
      for(int j=0;j<16;++j){ float2 v = er[j]; xr[j] = pack2bf(v.x, v.y); }
      // prefetch next timestep's chars2ix-driven embedding row (gfx1250 global_prefetch_b8)
      if(step < 15){
        int tn = dir ? (t - 1) : (t + 1);
        int cin = chars2ix[widx*16 + tn];
        __builtin_prefetch(char_emb + (size_t)cin*64 + half*32, 0, 1);
      }
      __syncthreads();

      // hoist A fragments (must read old h before writing new h)
      v16bf ax[2], ah[2];
#pragma unroll
      for(int kb=0;kb<2;++kb){
        ax[kb] = load_afrag(xt , 64, kb, lane);
        ah[kb] = load_afrag(hsh, 64, kb, lane);
      }
#pragma unroll
      for(int cg=0; cg<4; ++cg){
        v8f acc[4];
#pragma unroll
        for(int g=0;g<4;++g){
          int n = g*64 + cg*16 + (lane & 15);
          acc[g] = v8f_bcast(bih[n] + bhh[n]);
        }
#pragma unroll
        for(int kb=0;kb<2;++kb){
#pragma unroll
          for(int g=0;g<4;++g){
            int n0 = g*64 + cg*16;
            acc[g] = wmma_bf(ax[kb], load_bfrag(Wih + (size_t)n0*64, 64, kb, lane), acc[g]);
            acc[g] = wmma_bf(ah[kb], load_bfrag(Whh + (size_t)n0*64, 64, kb, lane), acc[g]);
          }
        }
#pragma unroll
        for(int e=0;e<8;++e){
          float ig = sigf(acc[0][e]);
          float fg = sigf(acc[1][e]);
          float gg = tanhf(acc[2][e]);
          float og = sigf(acc[3][e]);
          float cn = fg*cst[cg][e] + ig*gg;
          cst[cg][e] = cn;
          float h = og*tanhf(cn);
          int r = e + (half ? 8 : 0);
          int col = cg*16 + (lane & 15);
          hsh[r*64 + col] = f2bf(h);
          if(step == 15) hc[r*128 + dir*64 + col] = f2bf(h);   // final hidden
        }
      }
      __syncthreads();
    }
  }
  __syncthreads();
  // chvec = [hf|hb] @ char_out_W^T + b   (16x128 x 128x64)
#pragma unroll
  for(int nt=0; nt<4; ++nt){
    v8f acc = v8f_bcast(cout_b[nt*16 + (lane & 15)]);
#pragma unroll
    for(int kb=0; kb<4; ++kb){
      v16bf a = load_afrag(hc, 128, kb, lane);
      v16bf b = load_bfrag(coutW + (size_t)(nt*16)*128, 128, kb, lane);
      acc = wmma_bf(a, b, acc);
    }
#pragma unroll
    for(int e=0;e<8;++e){
      int m = word0 + e + ((lane & 16) ? 8 : 0);
      chvec[(size_t)m*64 + nt*16 + (lane & 15)] = acc[e];
    }
  }
}

// ---------- pack vec (bf16) and hcat tail cols ----------
__global__ void pack_vec_kernel(
    const int* __restrict__ words2ix, const float* __restrict__ word_emb,
    const float* __restrict__ chvec, const float* __restrict__ fv,
    u16* __restrict__ vecb, u16* __restrict__ hcatb)
{
  int i = blockIdx.x*blockDim.x + threadIdx.x;
  const int N1 = 16384*320;
  if(i < N1){
    int m = i/320, c = i%320;
    float v;
    if(c < 128)      v = word_emb[(size_t)words2ix[m]*128 + c];
    else if(c < 192) v = chvec[(size_t)m*64 + (c-128)];
    else             v = fv[(size_t)m*128 + (c-192)];
    vecb[i] = f2bf(v);
  } else {
    int j = i - N1;
    if(j < 16384*192){
      int m = j/192, c = j%192;
      float v = (c < 64) ? chvec[(size_t)m*64 + c] : fv[(size_t)m*128 + (c-64)];
      hcatb[(size_t)m*704 + 512 + c] = f2bf(v);
    }
  }
}

// ---------- sentence LSTM input GEMM: xW[t][b][1024] = vec @ Wih^T + bih + bhh ----------
__global__ __launch_bounds__(256,1) void lstm_in_gemm_kernel(
    const u16* __restrict__ vecb, const u16* __restrict__ Wih,
    const float* __restrict__ bih, const float* __restrict__ bhh,
    float* __restrict__ xW)
{
  int tid = threadIdx.x, wave = tid>>5, lane = tid&31;
  int task = blockIdx.x*8 + wave;       // 1024 mtiles * 64 ntiles
  int nt = task & 63, mt = task >> 6;
  int m0 = mt*16, n0 = nt*16;
  int coll = lane & 15, roff = (lane & 16) ? 8 : 0;
  const u16* abase = vecb + (size_t)m0*320 + (size_t)coll*320;
  const u16* bbase = Wih  + (size_t)n0*320 + (size_t)coll*320;
  v8f acc = v8f_bcast(bih[n0+coll] + bhh[n0+coll]);
#pragma unroll
  for(int kb=0; kb<10; ++kb){
    if(kb < 9){  // pull next K-block toward the WGP while this WMMA runs
      __builtin_prefetch(abase + (kb+1)*32, 0, 1);
      __builtin_prefetch(bbase + (kb+1)*32, 0, 1);
    }
    v16bf a = load_afrag(vecb + (size_t)m0*320, 320, kb, lane);
    v16bf b = load_bfrag(Wih  + (size_t)n0*320, 320, kb, lane);
    acc = wmma_bf(a, b, acc);
  }
#pragma unroll
  for(int e=0;e<8;++e){
    int m = m0 + e + roff;
    int bb = m >> 7, s = m & 127;       // vec rows are (b,s); output reindexed to [t][b]
    xW[((size_t)s*128 + bb)*1024 + n0 + coll] = acc[e];
  }
}

// ---------- sentence recurrent BiLSTM: one workgroup per direction, h in LDS, c in VGPRs ----------
__global__ __launch_bounds__(1024,1) void sent_lstm_kernel(
    const float* __restrict__ xW_f, const float* __restrict__ xW_b,
    const u16* __restrict__ Whh_f, const u16* __restrict__ Whh_b,
    u16* __restrict__ hcatb)
{
  __shared__ u16 hsh[128*256];          // h state, bf16, 64KB
  int tid = threadIdx.x, wave = tid>>5, lane = tid&31;
  int dir = blockIdx.x;
  const float* xW = dir ? xW_b : xW_f;
  const u16* Whh  = dir ? Whh_b : Whh_f;
  {
    u32* hz = (u32*)hsh;
    for(int i=tid; i<128*128; i+=1024) hz[i] = 0u;
  }
  __syncthreads();
  int mtile = wave >> 2;                // 0..7 (wave's 4 pairs share one mtile)
  int cg0 = (wave & 3) * 4;             // column groups cg0..cg0+3
  int m0 = mtile*16;
  int roff = (lane & 16) ? 8 : 0;
  int coll = lane & 15;
  v8f cst[4];
#pragma unroll
  for(int p=0;p<4;++p) cst[p] = v8f_bcast(0.f);
  float hv[4][8];

  for(int step=0; step<128; ++step){
    int t = dir ? (127 - step) : step;
    // prefetch next timestep's xW rows for this wave's tiles (one line per gate block)
    if(step < 127){
      int tn = dir ? (t - 1) : (t + 1);
#pragma unroll
      for(int g=0; g<4; ++g)
        __builtin_prefetch(xW + ((size_t)tn*128 + m0 + roff)*1024 + g*256 + cg0*16, 0, 1);
    }
#pragma unroll
    for(int p=0;p<4;++p){
      int cg = cg0 + p;
      v8f acc[4];
#pragma unroll
      for(int g=0;g<4;++g){
        int n = g*256 + cg*16 + coll;
#pragma unroll
        for(int e=0;e<8;++e)
          acc[g][e] = xW[((size_t)t*128 + m0 + e + roff)*1024 + n];
      }
#pragma unroll
      for(int kb=0;kb<8;++kb){
        v16bf a = load_afrag(hsh + (size_t)m0*256, 256, kb, lane);
#pragma unroll
        for(int g=0;g<4;++g){
          int n0 = g*256 + cg*16;
          acc[g] = wmma_bf(a, load_bfrag(Whh + (size_t)n0*256, 256, kb, lane), acc[g]);
        }
      }
#pragma unroll
      for(int e=0;e<8;++e){
        float ig = sigf(acc[0][e]);
        float fg = sigf(acc[1][e]);
        float gg = tanhf(acc[2][e]);
        float og = sigf(acc[3][e]);
        float cn = fg*cst[p][e] + ig*gg;
        cst[p][e] = cn;
        hv[p][e] = og*tanhf(cn);
      }
    }
    __syncthreads();                    // everyone done reading old h
#pragma unroll
    for(int p=0;p<4;++p){
      int cg = cg0 + p;
#pragma unroll
      for(int e=0;e<8;++e){
        int r = m0 + e + roff;          // batch index
        int col = cg*16 + coll;
        u16 hb = f2bf(hv[p][e]);
        hsh[(size_t)r*256 + col] = hb;
        hcatb[((size_t)r*128 + t)*704 + dir*256 + col] = hb;
      }
    }
    __syncthreads();                    // new h visible before next step
  }
}

// ---------- tag projection: feats[16384][32] = hcat @ tag_W^T + tag_b (pad -1e30) ----------
__global__ __launch_bounds__(256,1) void tag_gemm_kernel(
    const u16* __restrict__ hcatb, const u16* __restrict__ tagWb,
    const float* __restrict__ tag_b, float* __restrict__ feats)
{
  int tid = threadIdx.x, wave = tid>>5, lane = tid&31;
  int task = blockIdx.x*8 + wave;       // 1024 mtiles * 2 ntiles
  int nt = task & 1, mt = task >> 1;
  int m0 = mt*16, n0 = nt*16;
  int coll = lane & 15, roff = (lane & 16) ? 8 : 0;
  int n = n0 + coll;
  const u16* abase = hcatb + (size_t)m0*704 + (size_t)coll*704;
  v8f acc = v8f_bcast(n < 27 ? tag_b[n] : -1e30f);
#pragma unroll
  for(int kb=0; kb<22; ++kb){
    if(kb < 21) __builtin_prefetch(abase + (kb+1)*32, 0, 1);
    v16bf a = load_afrag(hcatb + (size_t)m0*704, 704, kb, lane);
    v16bf b = load_bfrag(tagWb + (size_t)n0*704, 704, kb, lane);
    acc = wmma_bf(a, b, acc);
  }
#pragma unroll
  for(int e=0;e<8;++e)
    feats[(size_t)(m0 + e + roff)*32 + n] = acc[e];
}

// ---------- argmax over tags ----------
__global__ void tag_argmax_kernel(const float* __restrict__ feats, float* __restrict__ out){
  int m = blockIdx.x*blockDim.x + threadIdx.x;
  if(m < 16384){
    const float* f = feats + (size_t)m*32;
    float best = f[0]; int arg = 0;
#pragma unroll
    for(int k=1;k<27;++k){ float v=f[k]; if(v>best){ best=v; arg=k; } }
    out[m] = (float)arg;
  }
}

// ---------- CRF Viterbi decode (K=27, START=25, STOP=26) ----------
__global__ void viterbi_kernel(
    const float* __restrict__ feats, const int* __restrict__ mask,
    const float* __restrict__ trans, float* __restrict__ out_score,
    float* __restrict__ out_path, int* __restrict__ bp)
{
  __shared__ float tr[27*27];
  int tid = threadIdx.x;
  for(int i=tid; i<27*27; i+=128) tr[i] = trans[i];
  __syncthreads();
  int b = tid;
  float sc[27];
#pragma unroll
  for(int k=0;k<27;++k) sc[k] = feats[(size_t)(b*128)*32 + k] + tr[25*27 + k];
  for(int s=1; s<128; ++s){
    int mt = mask[b*128 + s];
    float ns[27]; int bk[27];
#pragma unroll
    for(int k=0;k<27;++k){
      float best = -3.0e38f; int arg = 0;
#pragma unroll
      for(int j=0;j<27;++j){
        float v = sc[j] + tr[j*27 + k];
        if(v > best){ best = v; arg = j; }
      }
      if(mt){ ns[k] = best + feats[(size_t)(b*128 + s)*32 + k]; bk[k] = arg; }
      else  { ns[k] = sc[k]; bk[k] = k; }
    }
#pragma unroll
    for(int k=0;k<27;++k){
      sc[k] = ns[k];
      bp[((size_t)(s-1)*128 + b)*27 + k] = bk[k];
    }
  }
  float best = -3.0e38f; int last = 0;
#pragma unroll
  for(int k=0;k<27;++k){
    float v = sc[k] + tr[k*27 + 26];
    if(v > best){ best = v; last = k; }
  }
  out_score[b] = best;
  out_path[b*128 + 127] = (float)last;
  int tag = last;
  for(int s=126; s>=0; --s){
    tag = bp[((size_t)s*128 + b)*27 + tag];
    out_path[b*128 + s] = (float)tag;
  }
}

// ---------- host ----------
extern "C" void kernel_launch(void* const* d_in, const int* in_sizes, int n_in,
                              void* d_out, int out_size, void* d_ws, size_t ws_size,
                              hipStream_t stream){
  (void)in_sizes; (void)n_in; (void)out_size; (void)ws_size;
  const int*   words2ix = (const int*)  d_in[0];
  const int*   chars2ix = (const int*)  d_in[1];
  const float* features = (const float*)d_in[2];
  const int*   amask    = (const int*)  d_in[3];
  const float* word_emb = (const float*)d_in[4];
  const float* char_emb = (const float*)d_in[5];
  const float* cWih_f = (const float*)d_in[6];
  const float* cWhh_f = (const float*)d_in[7];
  const float* cbih_f = (const float*)d_in[8];
  const float* cbhh_f = (const float*)d_in[9];
  const float* cWih_b = (const float*)d_in[10];
  const float* cWhh_b = (const float*)d_in[11];
  const float* cbih_b = (const float*)d_in[12];
  const float* cbhh_b = (const float*)d_in[13];
  const float* coutW  = (const float*)d_in[14];
  const float* cout_b = (const float*)d_in[15];
  const float* featW  = (const float*)d_in[16];
  const float* featb  = (const float*)d_in[17];
  const float* lWih_f = (const float*)d_in[18];
  const float* lWhh_f = (const float*)d_in[19];
  const float* lbih_f = (const float*)d_in[20];
  const float* lbhh_f = (const float*)d_in[21];
  const float* lWih_b = (const float*)d_in[22];
  const float* lWhh_b = (const float*)d_in[23];
  const float* lbih_b = (const float*)d_in[24];
  const float* lbhh_b = (const float*)d_in[25];
  const float* tagW   = (const float*)d_in[26];
  const float* tagb   = (const float*)d_in[27];
  const float* trans  = (const float*)d_in[28];

  char* ws = (char*)d_ws; size_t off = 0;
  auto alloc = [&](size_t bytes)->char*{
    char* p = ws + off; off = (off + bytes + 255) & ~(size_t)255; return p;
  };
  float* chvec = (float*)alloc(16384ull*64*4);
  float* fv    = (float*)alloc(16384ull*128*4);
  u16*   vecb  = (u16*)  alloc(16384ull*320*2);
  u16*   hcatb = (u16*)  alloc(16384ull*704*2);
  float* xWf   = (float*)alloc(16384ull*1024*4);
  float* xWb   = (float*)alloc(16384ull*1024*4);
  float* feats = (float*)alloc(16384ull*32*4);
  int*   bp    = (int*)  alloc(127ull*128*27*4);
  u16* bcWih_f = (u16*)alloc(256*64*2);
  u16* bcWhh_f = (u16*)alloc(256*64*2);
  u16* bcWih_b = (u16*)alloc(256*64*2);
  u16* bcWhh_b = (u16*)alloc(256*64*2);
  u16* bcoutW  = (u16*)alloc(64*128*2);
  u16* blWih_f = (u16*)alloc(1024*320*2);
  u16* blWih_b = (u16*)alloc(1024*320*2);
  u16* blWhh_f = (u16*)alloc(1024*256*2);
  u16* blWhh_b = (u16*)alloc(1024*256*2);
  u16* btagW   = (u16*)alloc(32*704*2);

  auto cvt = [&](const float* s, u16* d, int n){
    cvt_bf16_kernel<<<(n+255)/256, 256, 0, stream>>>(s, d, n);
  };
  cvt(cWih_f, bcWih_f, 256*64);   cvt(cWhh_f, bcWhh_f, 256*64);
  cvt(cWih_b, bcWih_b, 256*64);   cvt(cWhh_b, bcWhh_b, 256*64);
  cvt(coutW,  bcoutW,  64*128);
  cvt(lWih_f, blWih_f, 1024*320); cvt(lWih_b, blWih_b, 1024*320);
  cvt(lWhh_f, blWhh_f, 1024*256); cvt(lWhh_b, blWhh_b, 1024*256);
  pad_tagw_kernel<<<(32*704+255)/256, 256, 0, stream>>>(tagW, btagW);

  feat_kernel<<<128, 256, 0, stream>>>(features, featW, featb, fv);
  char_lstm_kernel<<<256, 128, 0, stream>>>(chars2ix, char_emb,
      bcWih_f, bcWhh_f, cbih_f, cbhh_f,
      bcWih_b, bcWhh_b, cbih_b, cbhh_b,
      bcoutW, cout_b, chvec);
  {
    int n = 16384*(320+192);
    pack_vec_kernel<<<(n+255)/256, 256, 0, stream>>>(words2ix, word_emb, chvec, fv, vecb, hcatb);
  }
  lstm_in_gemm_kernel<<<8192, 256, 0, stream>>>(vecb, blWih_f, lbih_f, lbhh_f, xWf);
  lstm_in_gemm_kernel<<<8192, 256, 0, stream>>>(vecb, blWih_b, lbih_b, lbhh_b, xWb);
  sent_lstm_kernel<<<2, 1024, 0, stream>>>(xWf, xWb, blWhh_f, blWhh_b, hcatb);
  tag_gemm_kernel<<<256, 256, 0, stream>>>(hcatb, btagW, tagb, feats);

  float* out = (float*)d_out;
  tag_argmax_kernel<<<64, 256, 0, stream>>>(feats, out + 128 + 16384);
  viterbi_kernel<<<1, 128, 0, stream>>>(feats, amask, trans, out, out + 128, bp);
}
